// De_MoElayer_90512140795929
// MI455X (gfx1250) — matrix operand order repaired and checked by reading, not compiled, not run on hardware
//
#include <hip/hip_runtime.h>
#include <cstddef>
#include <cstdint>

typedef __attribute__((ext_vector_type(16))) _Float16 v16h;
typedef __attribute__((ext_vector_type(8)))  _Float16 v8h;
typedef __attribute__((ext_vector_type(8)))  float    v8f;

#define B_SZ  16
#define C_SZ  64
#define H_SZ  128
#define W_SZ  128
#define E_SZ  5
#define K_TOT 576          // 9 taps * 64 channels; k = tap*64 + c_in
#define HW    (H_SZ * W_SZ)
#define WP    130          // padded width (zero cols 0 and 129)
#define CP    72           // padded channel dim (row stride 144 B, bank-friendly)

// ---------------------------------------------------------------------------
// 1. Global average pool: one block per (b,c), mean over H*W contiguous floats
// ---------------------------------------------------------------------------
__global__ __launch_bounds__(256) void pool_kernel(const float* __restrict__ x,
                                                   float* __restrict__ pooled) {
  const int bc = blockIdx.x;                         // b*C + c, 0..1023
  const float4* src = (const float4*)(x + (size_t)bc * HW);
  float s = 0.f;
  for (int i = threadIdx.x; i < HW / 4; i += 256) {
    float4 v = src[i];
    s += v.x + v.y + v.z + v.w;
  }
#pragma unroll
  for (int off = 16; off > 0; off >>= 1) s += __shfl_xor(s, off);
  __shared__ float red[8];
  if ((threadIdx.x & 31) == 0) red[threadIdx.x >> 5] = s;
  __syncthreads();
  if (threadIdx.x == 0) {
    float t = 0.f;
#pragma unroll
    for (int i = 0; i < 8; ++i) t += red[i];
    pooled[bc] = t * (1.0f / HW);
  }
}

// ---------------------------------------------------------------------------
// 2. Noisy top-1 router: TOP_K==1 => gate weight is exactly 1.0 at argmax.
// ---------------------------------------------------------------------------
__global__ void router_kernel(const float* __restrict__ pooled,
                              const float* __restrict__ noise,
                              const float* __restrict__ w_gate,
                              const float* __restrict__ b_gate,
                              const float* __restrict__ w_noise,
                              const float* __restrict__ b_noise,
                              int* __restrict__ e_star) {
  const int b = threadIdx.x;
  if (b >= B_SZ) return;
  float best = -3.4e38f;
  int   bi   = 0;
  for (int e = 0; e < E_SZ; ++e) {
    float lg = b_gate[e], nl = b_noise[e];
    for (int c = 0; c < C_SZ; ++c) {
      float pc = pooled[b * C_SZ + c];
      lg += pc * w_gate[c * E_SZ + e];     // w_gate is [C,E]
      nl += pc * w_noise[c * E_SZ + e];
    }
    float sp = (nl > 20.f) ? nl : log1pf(expf(nl));  // softplus
    float v  = lg + noise[b * E_SZ + e] * sp;
    if (v > best) { best = v; bi = e; }
  }
  e_star[b] = bi;
}

// ---------------------------------------------------------------------------
// 3. Materialize the 5 transformed expert kernels as f16, layout [e][m][k],
//    k = tap*64 + c_in  (so a K-chunk of 32 has one fixed tap).
// ---------------------------------------------------------------------------
__global__ __launch_bounds__(256) void wmat_kernel(const float* __restrict__ w1,
                                                   const float* __restrict__ wcd,
                                                   const float* __restrict__ whd,
                                                   const float* __restrict__ wvd,
                                                   const float* __restrict__ wad,
                                                   _Float16* __restrict__ Wf16) {
  const int idx = blockIdx.x * 256 + threadIdx.x;
  if (idx >= E_SZ * C_SZ * K_TOT) return;
  const int k   = idx % K_TOT;
  const int m   = (idx / K_TOT) % C_SZ;
  const int e   = idx / (K_TOT * C_SZ);
  const int tap = k >> 6;          // 0..8
  const int i   = k & 63;          // input channel
  const int mi9 = (m * C_SZ + i) * 9;
  const int mi3 = (m * C_SZ + i) * 3;
  float val = 0.f;
  if (e == 0) {
    val = w1[mi9 + tap];
  } else if (e == 1) {                       // central-difference: w4 -= sum(all)
    float s = 0.f;
    for (int t = 0; t < 9; ++t) s += wcd[mi9 + t];
    val = wcd[mi9 + tap] - (tap == 4 ? s : 0.f);
  } else if (e == 2) {                       // horizontal diff: +{0,3,6}, -{2,5,8}
    const int c3 = tap % 3;
    const float wv = whd[mi3 + tap / 3];
    val = (c3 == 0) ? wv : (c3 == 2 ? -wv : 0.f);
  } else if (e == 3) {                       // vertical diff: +{0,1,2}, -{6,7,8}
    if (tap < 3)       val = wvd[mi3 + tap];
    else if (tap >= 6) val = -wvd[mi3 + tap - 6];
  } else {                                   // angular diff: w - THETA*w[perm]
    const int perm[9] = {3, 0, 1, 6, 4, 2, 7, 8, 5};
    val = wad[mi9 + tap] - 1.0f * wad[mi9 + perm[tap]];
  }
  Wf16[idx] = (_Float16)val;
}

// ---------------------------------------------------------------------------
// 4. Implicit-GEMM 3x3 conv via v_wmma_f32_16x16x32_f16.
//    One block per (b, output row PAIR). GEMM per row: M=64, N=128, K=576.
//    8 waves: wave w -> output row h0+(w>>2), pixel base (w&3)*32, and owns
//    TWO 16-pixel N-tiles so each A fragment feeds two WMMAs.
//    LDS patch (dynamic, 4 rows) is transposed + channel-contiguous:
//    patch[r][w+1][c] -> every B fragment is two ds_load_b128, no edge branches.
// ---------------------------------------------------------------------------
__device__ __forceinline__ v16h ld_frag16(const _Float16* p) {
  v8h lo = *(const v8h*)p;
  v8h hi = *(const v8h*)(p + 8);
  return __builtin_shufflevector(lo, hi,
           0, 1, 2, 3, 4, 5, 6, 7, 8, 9, 10, 11, 12, 13, 14, 15);
}

__global__ __launch_bounds__(256) void conv_wmma_kernel(
    const float* __restrict__ x, const _Float16* __restrict__ Wf16,
    const int* __restrict__ e_star,
    const float* __restrict__ b1, const float* __restrict__ bcd,
    const float* __restrict__ bhd, const float* __restrict__ bvd,
    const float* __restrict__ bad, float* __restrict__ y) {
  extern __shared__ _Float16 patch[];      // [4][WP][CP], 74880 B
  const int blk = blockIdx.x;
  const int b   = blk >> 6;                // example
  const int h0  = (blk & 63) << 1;         // first of two output rows
  const int e   = e_star[b];
  const _Float16* __restrict__ A = Wf16 + (size_t)e * C_SZ * K_TOT;
  const float* bias = (e == 0) ? b1 : (e == 1) ? bcd : (e == 2) ? bhd
                      : (e == 3) ? bvd : bad;
  const int tid = threadIdx.x;

  // Zero the two padded W columns (w=0 and w=WP-1) of all 4 rows.
  for (int z = tid; z < 2 * 4 * C_SZ; z += 256) {
    const int col = (z & 1) ? (WP - 1) : 0;
    const int rr  = (z >> 1) & 3;
    const int ii  = (z >> 1) >> 2;
    patch[(rr * WP + col) * CP + ii] = (_Float16)0.f;
  }

  // Stage x[b, :, h0-1..h0+2, :] transposed into LDS (consecutive lanes take
  // consecutive channels -> conflict-free 2-byte-contiguous LDS stores).
  for (int idx = tid; idx < C_SZ * 4 * (W_SZ / 4); idx += 256) {
    const int i  = idx & 63;            // channel
    const int rw = idx >> 6;
    const int w4 = rw & 31;             // group of 4 w's
    const int r  = rw >> 5;             // 0..3
    const int hh = h0 + r - 1;
    float4 v = {0.f, 0.f, 0.f, 0.f};
    if (hh >= 0 && hh < H_SZ)
      v = *(const float4*)(x + (((size_t)b * C_SZ + i) * H_SZ + hh) * W_SZ + w4 * 4);
    patch[(r * WP + w4 * 4 + 1) * CP + i] = (_Float16)v.x;
    patch[(r * WP + w4 * 4 + 2) * CP + i] = (_Float16)v.y;
    patch[(r * WP + w4 * 4 + 3) * CP + i] = (_Float16)v.z;
    patch[(r * WP + w4 * 4 + 4) * CP + i] = (_Float16)v.w;
  }
  __syncthreads();

  const int wave = tid >> 5;
  const int lane = tid & 31;
  const int n    = lane & 15;
  const int hi   = lane >> 4;         // lane half selects K sub-range
  const int hr   = wave >> 2;         // which of the two output rows
  const int h    = h0 + hr;
  const int nb   = (wave & 3) << 5;   // 32-pixel strip base
  const int p0   = nb + n;            // first N-tile pixel
  const int p1   = nb + 16 + n;       // second N-tile pixel
  const int aoff = hi << 3;           // unused; A fragment K sub-offset is coff (0 or 16)
  const int coff = hi << 4;           // B fragment channel sub-offset (0 or 16)

  v8f acc0[4] = {v8f{}, v8f{}, v8f{}, v8f{}};
  v8f acc1[4] = {v8f{}, v8f{}, v8f{}, v8f{}};
#pragma unroll
  for (int ch = 0; ch < 18; ++ch) {
    const int tap = ch >> 1;                 // constant after unroll
    const int dh  = tap / 3;
    const int dw  = tap % 3 - 1;
    const int i0  = ((ch & 1) << 5) + coff;  // starting channel of this lane's 16
    // B 32x16, this code's contiguous k order: lanes 0-15 hold K=0..15 (V0..7), lanes 16-31 K=16..31.
    const int rowoff = ((hr + dh) * WP + dw + 1) * CP + i0;
    v16h bf0 = ld_frag16(&patch[rowoff + p0 * CP]);
    v16h bf1 = ld_frag16(&patch[rowoff + p1 * CP]);
    // A 16x32 layout: lane holds row M=lane&15; lanes 0-15: K=0..15,
    // lanes 16-31: K=16..31 (same contiguous order as B).
#pragma unroll
    for (int mt = 0; mt < 4; ++mt) {
      const _Float16* arow = A + (size_t)(mt * 16 + n) * K_TOT + ch * 32;
      v8h a_lo = *(const v8h*)(arow + coff);
      v8h a_hi = *(const v8h*)(arow + coff + 8);
      v16h af  = __builtin_shufflevector(a_lo, a_hi,
                   0, 1, 2, 3, 4, 5, 6, 7, 8, 9, 10, 11, 12, 13, 14, 15);
      acc0[mt] = __builtin_amdgcn_wmma_f32_16x16x32_f16(
          false, af, false, bf0, (short)0, acc0[mt], false, false);
      acc1[mt] = __builtin_amdgcn_wmma_f32_16x16x32_f16(
          false, af, false, bf1, (short)0, acc1[mt], false, false);
    }
  }

  // D layout: VGPR r, lanes 0-15 -> (M=r, N=lane); lanes 16-31 -> (M=8+r).
#pragma unroll
  for (int mt = 0; mt < 4; ++mt) {
#pragma unroll
    for (int r = 0; r < 8; ++r) {
      const int m = mt * 16 + r + (hi << 3);
      const float bm = bias[m];
      float* dst = y + (((size_t)b * C_SZ + m) * H_SZ + h) * W_SZ;
      dst[p0] = acc0[mt][r] + bm;
      dst[p1] = acc1[mt][r] + bm;
    }
  }
}

// ---------------------------------------------------------------------------
// 5. In-place softmax over each example's flattened C*H*W (gate weight == 1).
// ---------------------------------------------------------------------------
__global__ __launch_bounds__(256) void smax_partial(const float* __restrict__ y,
                                                    float* __restrict__ pmax) {
  const int blk = blockIdx.x;                       // b*64 + chunk
  const float4* src = (const float4*)(y + (size_t)blk * 16384);
  float m = -3.4e38f;
  for (int i = threadIdx.x; i < 4096; i += 256) {
    float4 v = src[i];
    m = fmaxf(m, fmaxf(fmaxf(v.x, v.y), fmaxf(v.z, v.w)));
  }
#pragma unroll
  for (int off = 16; off > 0; off >>= 1) m = fmaxf(m, __shfl_xor(m, off));
  __shared__ float red[8];
  if ((threadIdx.x & 31) == 0) red[threadIdx.x >> 5] = m;
  __syncthreads();
  if (threadIdx.x == 0) {
    float t = red[0];
    for (int i = 1; i < 8; ++i) t = fmaxf(t, red[i]);
    pmax[blk] = t;
  }
}

__global__ void finalize_max(const float* __restrict__ pmax, float* __restrict__ bmax) {
  const int b = threadIdx.x;
  if (b >= B_SZ) return;
  float m = -3.4e38f;
  for (int i = 0; i < 64; ++i) m = fmaxf(m, pmax[b * 64 + i]);
  bmax[b] = m;
}

__global__ __launch_bounds__(256) void ssum_partial(const float* __restrict__ y,
                                                    const float* __restrict__ bmax,
                                                    float* __restrict__ psum) {
  const int blk = blockIdx.x;
  const float mb = bmax[blk >> 6];
  const float4* src = (const float4*)(y + (size_t)blk * 16384);
  float s = 0.f;
  for (int i = threadIdx.x; i < 4096; i += 256) {
    float4 v = src[i];
    s += expf(v.x - mb) + expf(v.y - mb) + expf(v.z - mb) + expf(v.w - mb);
  }
#pragma unroll
  for (int off = 16; off > 0; off >>= 1) s += __shfl_xor(s, off);
  __shared__ float red[8];
  if ((threadIdx.x & 31) == 0) red[threadIdx.x >> 5] = s;
  __syncthreads();
  if (threadIdx.x == 0) {
    float t = 0.f;
    for (int i = 0; i < 8; ++i) t += red[i];
    psum[blk] = t;
  }
}

__global__ void finalize_sum(const float* __restrict__ psum, float* __restrict__ binv) {
  const int b = threadIdx.x;
  if (b >= B_SZ) return;
  float s = 0.f;
  for (int i = 0; i < 64; ++i) s += psum[b * 64 + i];
  binv[b] = 1.0f / s;
}

__global__ __launch_bounds__(256) void normalize_kernel(float* __restrict__ y,
                                                        const float* __restrict__ bmax,
                                                        const float* __restrict__ binv) {
  const int blk = blockIdx.x;
  const float mb = bmax[blk >> 6];
  const float sc = binv[blk >> 6];
  float4* dst = (float4*)(y + (size_t)blk * 16384);
  for (int i = threadIdx.x; i < 4096; i += 256) {
    float4 v = dst[i];
    v.x = expf(v.x - mb) * sc;
    v.y = expf(v.y - mb) * sc;
    v.z = expf(v.z - mb) * sc;
    v.w = expf(v.w - mb) * sc;
    dst[i] = v;
  }
}

// ---------------------------------------------------------------------------
extern "C" void kernel_launch(void* const* d_in, const int* in_sizes, int n_in,
                              void* d_out, int out_size, void* d_ws, size_t ws_size,
                              hipStream_t stream) {
  (void)in_sizes; (void)n_in; (void)out_size; (void)ws_size;
  const float* x       = (const float*)d_in[0];
  const float* noise   = (const float*)d_in[1];
  const float* w_gate  = (const float*)d_in[2];
  const float* b_gate  = (const float*)d_in[3];
  const float* w_noise = (const float*)d_in[4];
  const float* b_noise = (const float*)d_in[5];
  const float* w1      = (const float*)d_in[6];
  const float* b1      = (const float*)d_in[7];
  const float* w_cd    = (const float*)d_in[8];
  const float* b_cd    = (const float*)d_in[9];
  const float* w_hd    = (const float*)d_in[10];
  const float* b_hd    = (const float*)d_in[11];
  const float* w_vd    = (const float*)d_in[12];
  const float* b_vd    = (const float*)d_in[13];
  const float* w_ad    = (const float*)d_in[14];
  const float* b_ad    = (const float*)d_in[15];
  float* out = (float*)d_out;                 // conv output lives here, softmaxed in place

  char* ws = (char*)d_ws;
  size_t o = 0;
  _Float16* Wf16 = (_Float16*)(ws + o); o += (size_t)E_SZ * C_SZ * K_TOT * 2;  // 368640
  float* pooled  = (float*)(ws + o);    o += 1024 * 4;
  int*   e_star  = (int*)(ws + o);      o += 64;
  float* pmax    = (float*)(ws + o);    o += 1024 * 4;
  float* bmax    = (float*)(ws + o);    o += 64;
  float* psum    = (float*)(ws + o);    o += 1024 * 4;
  float* binv    = (float*)(ws + o);    o += 64;

  pool_kernel<<<B_SZ * C_SZ, 256, 0, stream>>>(x, pooled);
  router_kernel<<<1, 32, 0, stream>>>(pooled, noise, w_gate, b_gate, w_noise, b_noise, e_star);
  wmat_kernel<<<(E_SZ * C_SZ * K_TOT + 255) / 256, 256, 0, stream>>>(w1, w_cd, w_hd, w_vd, w_ad, Wf16);
  const size_t patch_bytes = (size_t)4 * WP * CP * sizeof(_Float16);   // 74880
  conv_wmma_kernel<<<B_SZ * (H_SZ / 2), 256, patch_bytes, stream>>>(
      x, Wf16, e_star, b1, b_cd, b_hd, b_vd, b_ad, out);
  smax_partial<<<B_SZ * 64, 256, 0, stream>>>(out, pmax);
  finalize_max<<<1, 32, 0, stream>>>(pmax, bmax);
  ssum_partial<<<B_SZ * 64, 256, 0, stream>>>(out, bmax, psum);
  finalize_sum<<<1, 32, 0, stream>>>(psum, binv);
  normalize_kernel<<<B_SZ * 64, 256, 0, stream>>>(out, bmax, binv);
}
